// ConditionalVariationalModule_75299366633595
// MI455X (gfx1250) — matrix operand order repaired and checked by reading, not compiled, not run on hardware
//
#include <hip/hip_runtime.h>

typedef __attribute__((ext_vector_type(16))) __bf16 v16bf;
typedef __attribute__((ext_vector_type(8)))  __bf16 v8bf;
typedef __attribute__((ext_vector_type(4)))  __bf16 v4bf;
typedef __attribute__((ext_vector_type(8)))  float  v8f;

#define INPUT_DIM  256
#define LATENT_DIM 128
#define HIDDEN_DIM 256
#define BATCH      1024
#define SEQ        256
#define BM         32          // batch rows per workgroup
#define XROW       640         // [ h(0:256) | z(256:384) | h(384:640) ]

// ---------------------------------------------------------------- helpers
__device__ __forceinline__ __bf16 f2bf(float f) {
  union { float f; unsigned u; } v; v.f = f;
  unsigned r = v.u + 0x7FFFu + ((v.u >> 16) & 1u);   // round-to-nearest-even
  union { unsigned short s; __bf16 b; } o; o.s = (unsigned short)(r >> 16);
  return o.b;
}

__device__ __forceinline__ unsigned short f2bf_bits(float f) {
  union { float f; unsigned u; } v; v.f = f;
  unsigned r = v.u + 0x7FFFu + ((v.u >> 16) & 1u);
  return (unsigned short)(r >> 16);
}

// A zero the compiler cannot see through. Added to weight pointers inside the
// timestep loop it defeats LICM (no mass-hoist of loop-invariant weight
// fragments -> no VGPR blowup/spills) while keeping pointer provenance, so
// loads stay global_load_b128 (LOADcnt only), not flat_load (LOADcnt+DScnt).
__device__ __forceinline__ int opaque_zero() {
  int x = 0;
  asm volatile("" : "+s"(x));
  return x;
}

// A fragment (16x32 bf16, M x K) from row-major LDS activations.
// lane<16:  e0..15 -> k=0..15   (row = lane)
// lane>=16: e0..15 -> k=16..31  (row = lane-16)
__device__ __forceinline__ v16bf load_a_frag(const __bf16* A, int rowlen,
                                             int m_tile, int kt, int lo, int hi) {
  const __bf16* p = A + (m_tile * 16 + lo) * rowlen + kt * 32 + hi * 16;
  v8bf c0 = *(const v8bf*)p;
  v8bf c1 = *(const v8bf*)(p + 8);
  v16bf a;
#pragma unroll
  for (int i = 0; i < 8; ++i) { a[i] = c0[i]; a[i + 8] = c1[i]; }
  return a;
}

// B fragment (32x16 bf16, K x N) from fragment-major packed weights:
// lane l holds column n = l&15, k-range (l>=16 ? 16..31 : 0..15), 16 contiguous halfs.
__device__ __forceinline__ v16bf load_b_frag(const unsigned short* Wp,
                                             int kt, int nt, int lane) {
  const __bf16* p = (const __bf16*)(Wp + ((((kt * 16 + nt) * 32) + lane) << 4));
  v8bf c0 = *(const v8bf*)p;
  v8bf c1 = *(const v8bf*)(p + 8);
  v16bf b;
#pragma unroll
  for (int i = 0; i < 8; ++i) { b[i] = c0[i]; b[i + 8] = c1[i]; }
  return b;
}

// Accumulate [16 x 32*ktiles] x [32*ktiles x 64] into 4 f32 accumulators.
__device__ __forceinline__ void gemm_acc(const __bf16* A, int rowlen, int ktiles,
                                         const unsigned short* Wp, v8f acc[4],
                                         int m_tile, int n_base, int lane,
                                         int lo, int hi) {
#pragma unroll
  for (int kt = 0; kt < ktiles; ++kt) {
    v16bf a = load_a_frag(A, rowlen, m_tile, kt, lo, hi);
#pragma unroll
    for (int i = 0; i < 4; ++i) {
      v16bf b = load_b_frag(Wp, kt, n_base + i, lane);
      acc[i] = __builtin_amdgcn_wmma_f32_16x16x32_bf16(
          false, a, false, b, (short)0, acc[i], false, false);
    }
  }
}

// bias + ReLU + bf16 store into LDS hidden buffer [BM][256]
__device__ __forceinline__ void store_hidden(__bf16* out, const float* bias,
                                             v8f acc[4], int m_tile, int n_base,
                                             int lo, int hi) {
#pragma unroll
  for (int i = 0; i < 4; ++i) {
    int nn = (n_base + i) * 16 + lo;
    float bv = bias[nn];
#pragma unroll
    for (int r = 0; r < 8; ++r) {
      int mm = m_tile * 16 + hi * 8 + r;
      float v = acc[i][r] + bv;
      out[mm * HIDDEN_DIM + nn] = f2bf(v > 0.f ? v : 0.f);
    }
  }
}

// ------------------------------------------------- weight packing (f32 -> bf16 frags)
__global__ void pack_weights(const float* __restrict__ W,
                             unsigned short* __restrict__ dst, int K) {
  int idx = blockIdx.x * blockDim.x + threadIdx.x;
  if (idx >= K * 256) return;
  int k = idx >> 8;
  int n = idx & 255;
  int ktile = k >> 5, kr = k & 31;
  int lane  = ((kr >> 4) << 4) | (n & 15);
  int e     = kr & 15;
  int ntile = n >> 4;
  int d = ((((ktile * 16 + ntile) * 32) + lane) << 4) + e;
  dst[d] = f2bf_bits(W[k * 256 + n]);
}

// ------------------------------------------------------------------ main kernel
__global__ __launch_bounds__(256) void cvae_scan_kernel(
    const float* __restrict__ enc, const float* __restrict__ prev_latent,
    const float* __restrict__ eps,
    const unsigned short* __restrict__ pw1g, const float* __restrict__ pb1,
    const unsigned short* __restrict__ pw2g, const float* __restrict__ pb2,
    const unsigned short* __restrict__ pw3g, const float* __restrict__ pb3,
    const unsigned short* __restrict__ qw1g, const float* __restrict__ qb1,
    const unsigned short* __restrict__ qw2g, const float* __restrict__ qb2,
    const unsigned short* __restrict__ qw3g, const float* __restrict__ qb3,
    float* __restrict__ out) {
  extern __shared__ char smem[];
  __bf16* xin  = (__bf16*)smem;                 // [BM][640] bf16
  __bf16* h1   = xin + BM * XROW;               // [BM][256] bf16
  __bf16* h2   = h1 + BM * HIDDEN_DIM;          // [BM][256] bf16
  float*  post = (float*)(h2 + BM * HIDDEN_DIM);// [BM][256] f32 (qm|qlv)

  const int b0   = blockIdx.x * BM;
  const int tid  = threadIdx.x;
  const int w    = tid >> 5;
  const int lane = tid & 31;
  const int lo   = lane & 15, hi = lane >> 4;
  const int m_tile = w & 1;          // 2 M-tiles of 16 rows
  const int n_base = (w >> 1) * 4;   // 4 N-tiles of 16 cols per wave

  const size_t OSTRIDE = (size_t)BATCH * SEQ * LATENT_DIM;
  float* zout   = out;
  float* pmout  = out + OSTRIDE;
  float* plvout = out + 2 * OSTRIDE;
  float* qmout  = out + 3 * OSTRIDE;
  float* qlvout = out + 4 * OSTRIDE;

  // init z state from prev_latent
  for (int i = tid; i < BM * LATENT_DIM; i += 256) {
    int r = i >> 7, j = i & 127;
    xin[r * XROW + INPUT_DIM + j] = f2bf(prev_latent[(size_t)(b0 + r) * LATENT_DIM + j]);
  }
  __syncthreads();

  for (int t = 0; t < SEQ; ++t) {
    // per-iteration opaque zero: weight addresses are not loop-invariant, so
    // B-fragments stream from L2 each step instead of being hoisted/spilled.
    const int z0 = opaque_zero();
    const unsigned short* pw1 = pw1g + z0;
    const unsigned short* pw2 = pw2g + z0;
    const unsigned short* pw3 = pw3g + z0;
    const unsigned short* qw1 = qw1g + z0;
    const unsigned short* qw2 = qw2g + z0;
    const unsigned short* qw3 = qw3g + z0;

    // ---- stage encoder features (both h slots of [h|z|h]), f32 -> bf16
    for (int i = tid; i < (BM * INPUT_DIM) / 4; i += 256) {
      int r = i >> 6, c4 = (i & 63) << 2;
      const float* src = &enc[((size_t)(b0 + r) * SEQ + t) * INPUT_DIM + c4];
      const float4 f = *(const float4*)src;
      if (t + 1 < SEQ)  // warm the cache for the next serial step
        __builtin_prefetch(src + INPUT_DIM, 0, 3);
      v4bf hv; hv[0] = f2bf(f.x); hv[1] = f2bf(f.y); hv[2] = f2bf(f.z); hv[3] = f2bf(f.w);
      *(v4bf*)&xin[r * XROW + c4]       = hv;
      *(v4bf*)&xin[r * XROW + 384 + c4] = hv;
    }
    __syncthreads();

    // ---- prior layer 1: [BM x 384] x [384 x 256] -> h1 (ReLU)
    {
      v8f acc[4] = {};
      gemm_acc(xin, XROW, 12, pw1, acc, m_tile, n_base, lane, lo, hi);
      store_hidden(h1, pb1, acc, m_tile, n_base, lo, hi);
    }
    __syncthreads();
    // ---- prior layer 2: [BM x 256] x [256 x 256] -> h2 (ReLU)
    {
      v8f acc[4] = {};
      gemm_acc(h1, HIDDEN_DIM, 8, pw2, acc, m_tile, n_base, lane, lo, hi);
      store_hidden(h2, pb2, acc, m_tile, n_base, lo, hi);
    }
    __syncthreads();
    // ---- prior layer 3 -> global pm | plv (no shared writes)
    {
      v8f acc[4] = {};
      gemm_acc(h2, HIDDEN_DIM, 8, pw3, acc, m_tile, n_base, lane, lo, hi);
#pragma unroll
      for (int i = 0; i < 4; ++i) {
        int nn = (n_base + i) * 16 + lo;
        float bv = pb3[nn];
        float* dst = (nn < 128) ? pmout : plvout;
        int col = nn & 127;
#pragma unroll
        for (int r = 0; r < 8; ++r) {
          int mm = m_tile * 16 + hi * 8 + r;
          dst[((size_t)(b0 + mm) * SEQ + t) * LATENT_DIM + col] = acc[i][r] + bv;
        }
      }
    }
    // ---- posterior layer 1: [BM x 640] x [640 x 256] -> h1 (ReLU)
    {
      v8f acc[4] = {};
      gemm_acc(xin, XROW, 20, qw1, acc, m_tile, n_base, lane, lo, hi);
      store_hidden(h1, qb1, acc, m_tile, n_base, lo, hi);
    }
    __syncthreads();
    // ---- posterior layer 2 -> h2 (ReLU)
    {
      v8f acc[4] = {};
      gemm_acc(h1, HIDDEN_DIM, 8, qw2, acc, m_tile, n_base, lane, lo, hi);
      store_hidden(h2, qb2, acc, m_tile, n_base, lo, hi);
    }
    __syncthreads();
    // ---- posterior layer 3 -> post (f32 LDS, qm|qlv)
    {
      v8f acc[4] = {};
      gemm_acc(h2, HIDDEN_DIM, 8, qw3, acc, m_tile, n_base, lane, lo, hi);
#pragma unroll
      for (int i = 0; i < 4; ++i) {
        int nn = (n_base + i) * 16 + lo;
        float bv = qb3[nn];
#pragma unroll
        for (int r = 0; r < 8; ++r) {
          int mm = m_tile * 16 + hi * 8 + r;
          post[mm * HIDDEN_DIM + nn] = acc[i][r] + bv;
        }
      }
    }
    __syncthreads();
    // ---- reparameterization + outputs: z = qm + eps * exp(0.5*qlv)
    for (int i = tid; i < BM * LATENT_DIM; i += 256) {
      int r = i >> 7, j = i & 127;
      float qm  = post[r * HIDDEN_DIM + j];
      float qlv = post[r * HIDDEN_DIM + 128 + j];
      size_t o  = ((size_t)(b0 + r) * SEQ + t) * LATENT_DIM + j;
      float zv  = qm + eps[o] * __expf(0.5f * qlv);
      zout[o] = zv; qmout[o] = qm; qlvout[o] = qlv;
      xin[r * XROW + INPUT_DIM + j] = f2bf(zv);   // state for next step
    }
    __syncthreads();
  }
}

// ------------------------------------------------------------------ launch
extern "C" void kernel_launch(void* const* d_in, const int* in_sizes, int n_in,
                              void* d_out, int out_size, void* d_ws, size_t ws_size,
                              hipStream_t stream) {
  const float* enc  = (const float*)d_in[0];
  const float* prev = (const float*)d_in[1];
  const float* eps  = (const float*)d_in[2];
  const float* pW1 = (const float*)d_in[3];  const float* pb1 = (const float*)d_in[4];
  const float* pW2 = (const float*)d_in[5];  const float* pb2 = (const float*)d_in[6];
  const float* pW3 = (const float*)d_in[7];  const float* pb3 = (const float*)d_in[8];
  const float* qW1 = (const float*)d_in[9];  const float* qb1 = (const float*)d_in[10];
  const float* qW2 = (const float*)d_in[11]; const float* qb2 = (const float*)d_in[12];
  const float* qW3 = (const float*)d_in[13]; const float* qb3 = (const float*)d_in[14];

  unsigned short* ws  = (unsigned short*)d_ws;
  unsigned short* wp1 = ws;                      // 384*256
  unsigned short* wp2 = wp1 + 384 * 256;         // 256*256
  unsigned short* wp3 = wp2 + 256 * 256;
  unsigned short* wq1 = wp3 + 256 * 256;         // 640*256
  unsigned short* wq2 = wq1 + 640 * 256;
  unsigned short* wq3 = wq2 + 256 * 256;

  pack_weights<<<(384 * 256 + 255) / 256, 256, 0, stream>>>(pW1, wp1, 384);
  pack_weights<<<(256 * 256 + 255) / 256, 256, 0, stream>>>(pW2, wp2, 256);
  pack_weights<<<(256 * 256 + 255) / 256, 256, 0, stream>>>(pW3, wp3, 256);
  pack_weights<<<(640 * 256 + 255) / 256, 256, 0, stream>>>(qW1, wq1, 640);
  pack_weights<<<(256 * 256 + 255) / 256, 256, 0, stream>>>(qW2, wq2, 256);
  pack_weights<<<(256 * 256 + 255) / 256, 256, 0, stream>>>(qW3, wq3, 256);

  const size_t SMEM = (size_t)BM * XROW * 2           // xin bf16
                    + 2 * (size_t)BM * HIDDEN_DIM * 2 // h1, h2 bf16
                    + (size_t)BM * HIDDEN_DIM * 4;    // post f32
  (void)hipFuncSetAttribute(reinterpret_cast<const void*>(cvae_scan_kernel),
                            hipFuncAttributeMaxDynamicSharedMemorySize, (int)SMEM);
  cvae_scan_kernel<<<BATCH / BM, 256, SMEM, stream>>>(
      enc, prev, eps, wp1, pb1, wp2, pb2, wp3, pb3,
      wq1, qb1, wq2, qb2, wq3, qb3, (float*)d_out);
}
